// SigLIPAttention_85358180041446
// MI455X (gfx1250) — matrix operand order repaired and checked by reading, not compiled, not run on hardware
//
#include <hip/hip_runtime.h>

typedef __bf16 bf16;
typedef __attribute__((ext_vector_type(16))) __bf16 v16bf;
typedef __attribute__((ext_vector_type(8)))  __bf16 v8bf;
typedef __attribute__((ext_vector_type(8)))  float  v8f;
typedef __attribute__((ext_vector_type(4)))  unsigned int v4u;
typedef __attribute__((ext_vector_type(8)))  unsigned int v8u;

static __device__ __forceinline__ v8f wmma_bf16(v16bf a, v16bf b, v8f c) {
  // D = A(16x32 bf16) * B(32x16 bf16) + C(16x16 f32)
  return __builtin_amdgcn_wmma_f32_16x16x32_bf16(false, a, false, b, (short)0, c,
                                                 false, false);
}

// ---------------------------------------------------------------------------
// TDM: issue a 2-D tile load (tile_d1 rows x tile_d0 elements, bf16) from
// global memory into LDS via tensor_load_to_lds. Descriptor per
// CDNA5 ISA:
//   group0: [1:0]=count=1, [63:32]=lds_addr, [120:64]=global_addr, [127:126]=2
//   group1: [17:16]=data_size(1 -> 2B), [79:48]=tensor_dim0, [111:80]=dim1,
//           [127:112]=tile_dim0, [143:128]=tile_dim1, [207:160]=dim0_stride
// Groups 2/3 omitted (2-D tensor). Completion tracked on TENSORcnt.
// ---------------------------------------------------------------------------
static __device__ __forceinline__ void tdm_load_2d(unsigned int lds_off,
                                                   const void* gaddr,
                                                   unsigned int row_stride_elems,
                                                   unsigned int tile_d0,
                                                   unsigned int tile_d1) {
  unsigned long long ga = (unsigned long long)gaddr;
  const unsigned int td0 = 0x7FFFFFu;  // huge tensor dims: tiles always in-bounds
  const unsigned int td1 = 0x7FFFFFu;
  v4u g0;
  g0[0] = 1u;                                                   // count=1
  g0[1] = lds_off;                                              // lds_addr
  g0[2] = (unsigned int)(ga & 0xFFFFFFFFu);                     // gaddr[31:0]
  g0[3] = (unsigned int)((ga >> 32) & 0x1FFFFFFu) | (2u << 30); // gaddr[56:32]|type
  v8u g1;
  g1[0] = (1u << 16);                          // data_size = 1 (2 bytes/elem)
  g1[1] = (td0 & 0xFFFFu) << 16;               // tensor_dim0[15:0] @ bits 63:48
  g1[2] = (td0 >> 16) | ((td1 & 0xFFFFu) << 16);
  g1[3] = (td1 >> 16) | (tile_d0 << 16);       // tile_dim0
  g1[4] = tile_d1 & 0xFFFFu;                   // tile_dim1 (tile_dim2 = 0)
  g1[5] = row_stride_elems;                    // tensor_dim0_stride[31:0]
  g1[6] = 0u;
  g1[7] = 0u;
  asm volatile("tensor_load_to_lds %0, %1" ::"s"(g0), "s"(g1) : "memory");
}

// ---------------------------------------------------------------------------
// Kernel 1: fp32 -> bf16 cast
// ---------------------------------------------------------------------------
__global__ __launch_bounds__(256) void cast_f32_to_bf16(
    const float* __restrict__ in, bf16* __restrict__ out, int n) {
  int i = blockIdx.x * 256 + threadIdx.x;
  if (i < n) out[i] = (bf16)in[i];
}

// ---------------------------------------------------------------------------
// Kernel 2/5: C[M,N] = A[M,K] * W[N,K]^T (+bias), bf16 in, f32 accumulate.
// TDM double-buffered, deep stages: block tile 16(M) x 512(N), 8 waves, each
// wave owns a 16x64 output strip (4 WMMA N-tiles). Each pipeline stage is a
// K=64 slice: per stage every wave TDMs its 64x64 slice of W (8 KB DMA) into
// LDS, wave 0 also TDMs the 16x64 A tile. 132 KB LDS total (two stages) --
// sized for CDNA5's 320 KB WGP LDS. Between barriers each wave runs 8 WMMAs
// (2 K-steps x 4 N-tiles), so sync cost is amortized 2x vs a K=32 stage.
// Fragment maps (contiguous k, not ISA table order): A 16x32: lane l: row=l%16, contiguous K[16h..+15].
// B 32x16: lane l: col=l%16, contiguous K[16h..+15].
// ---------------------------------------------------------------------------
template <bool OUT_BF16, bool HAS_BIAS>
__global__ __launch_bounds__(256) void gemm_wmma_tdm(
    const bf16* __restrict__ A, const bf16* __restrict__ W, void* __restrict__ Out,
    const float* __restrict__ bias, int M, int N, int K) {
  __shared__ bf16 sW[2][512 * 64];  // 64 KB per stage
  __shared__ bf16 sA[2][16 * 64];   //  2 KB per stage

  const int lane = threadIdx.x & 31;
  const int wave = threadIdx.x >> 5;
  const int half = lane >> 4;
  const int j    = lane & 15;
  const int m0   = blockIdx.y * 16;
  const int nblk = blockIdx.x * 512;
  const int n0   = nblk + wave * 64;

  v8f acc[4] = {};

  auto issue = [&](int p, int kk) {
    // each wave DMAs its own 64 rows x 64 k of the W tile (8 KB)
    tdm_load_2d((unsigned int)(uintptr_t)&sW[p][wave * 64 * 64],
                W + (size_t)(nblk + wave * 64) * K + kk, (unsigned)K, 64, 64);
    if (wave == 0) {  // one wave DMAs the shared A tile (2 KB)
      tdm_load_2d((unsigned int)(uintptr_t)&sA[p][0],
                  A + (size_t)m0 * K + kk, (unsigned)K, 64, 16);
    }
  };

  issue(0, 0);
  int p = 0;
  for (int kk = 0; kk < K; kk += 64) {
    const bool more = (kk + 64) < K;
    if (more) issue(p ^ 1, kk + 64);
    if (more) {
      if (wave == 0) __builtin_amdgcn_s_wait_tensorcnt(2);  // keep next in flight
      else           __builtin_amdgcn_s_wait_tensorcnt(1);
    } else {
      __builtin_amdgcn_s_wait_tensorcnt(0);
    }
    __syncthreads();  // all slices of stage p visible to all waves

#pragma unroll
    for (int s = 0; s < 2; ++s) {  // two K=32 sub-steps within the stage
      union { v16bf v; v8bf h[2]; } ua;
      const bf16* ap = &sA[p][0] + j * 64 + s * 32;
      ua.h[0] = *(const v8bf*)(ap + 16 * half);
      ua.h[1] = *(const v8bf*)(ap + 8 + 16 * half);
#pragma unroll
      for (int c = 0; c < 4; ++c) {
        const bf16* wp =
            &sW[p][0] + (wave * 64 + c * 16 + j) * 64 + s * 32 + 16 * half;
        acc[c] = wmma_bf16(ua.v, *(const v16bf*)wp, acc[c]);
      }
    }
    __syncthreads();  // everyone done with stage p before it is overwritten
    p ^= 1;
  }

#pragma unroll
  for (int c = 0; c < 4; ++c) {
    const int n = n0 + c * 16 + j;
    float bv = HAS_BIAS ? bias[n] : 0.0f;
#pragma unroll
    for (int r = 0; r < 8; ++r) {
      const int row = m0 + r + 8 * half;  // C-layout: VGPR r -> row r(+8)
      float v = acc[c][r] + bv;
      if (OUT_BF16) ((bf16*)Out)[(size_t)row * N + n] = (bf16)v;
      else          ((float*)Out)[(size_t)row * N + n] = v;
    }
  }
}

// ---------------------------------------------------------------------------
// Kernel 3: RoPE on q,k (token 0 = cls, untouched) + repack for attention:
//   qh, kh : [B*H, 1056, 64]  (rows >= 1025 zero-padded)
//   vt     : [B*H, 64, 1056]  (V transposed so P*V B-operand is contiguous)
// qkv GEMM output layout: [B*1025, 3*1024], col = which*1024 + h*64 + d.
// ---------------------------------------------------------------------------
__global__ __launch_bounds__(256) void rope_pack(
    const bf16* __restrict__ qkv, const float* __restrict__ sinp,
    const float* __restrict__ cosp, bf16* __restrict__ qh, bf16* __restrict__ kh,
    bf16* __restrict__ vt) {
  const int LP = 1056;
  int idx = blockIdx.x * 256 + threadIdx.x;
  int d   = idx & 63;
  int row = (idx >> 6) % LP;
  int bh  = idx / (LP * 64);
  if (bh >= 256) return;
  int b = bh >> 4, h = bh & 15;

  float qv = 0.f, kv = 0.f, vv = 0.f;
  if (row < 1025) {
    size_t base = (size_t)(b * 1025 + row) * 3072 + h * 64;
    qv = (float)qkv[base + d];
    kv = (float)qkv[base + 1024 + d];
    vv = (float)qkv[base + 2048 + d];
    if (row >= 1) {  // patch tokens get RoPE with sin/cos row (row-1)
      float s  = sinp[(row - 1) * 64 + d];
      float c  = cosp[(row - 1) * 64 + d];
      int   dp = (d < 32) ? d + 32 : d - 32;
      float qr = (float)qkv[base + dp];
      float kr = (float)qkv[base + 1024 + dp];
      if (d < 32) { qr = -qr; kr = -kr; }  // rotate_half
      qv = qv * c + qr * s;
      kv = kv * c + kr * s;
    }
  }
  size_t o = ((size_t)bh * LP + row) * 64 + d;
  qh[o] = (bf16)qv;
  kh[o] = (bf16)kv;
  vt[((size_t)bh * 64 + d) * LP + row] = (bf16)vv;
}

// ---------------------------------------------------------------------------
// Kernel 4: flash attention. One block per (b,h); 8 waves; wave w handles
// query tiles {w, w+8, ...} of the 65 tiles. Keys streamed in pairs of 16-row
// tiles so the P(16x32)*V(32x64) WMMA runs at K=32. Online softmax stats per
// row (8 rows per half-wave) with 16-lane shfl_xor reductions. P is converted
// from C-layout to A-layout through a per-wave LDS slab.
// ---------------------------------------------------------------------------
__global__ __launch_bounds__(256) void flash_attn(
    const bf16* __restrict__ qh, const bf16* __restrict__ kh,
    const bf16* __restrict__ vt, bf16* __restrict__ ao) {
  __shared__ bf16 lds_p[8][16 * 32];
  const int  lane = threadIdx.x & 31;
  const int  wave = threadIdx.x >> 5;
  const int  half = lane >> 4;
  const int  j    = lane & 15;
  const int  bh   = blockIdx.x;
  const int  b    = bh >> 4, h = bh & 15;
  const int  LP   = 1056;
  const bf16* qb = qh + (size_t)bh * LP * 64;
  const bf16* kb = kh + (size_t)bh * LP * 64;
  const bf16* vb = vt + (size_t)bh * 64 * LP;
  bf16* slab = &lds_p[wave][0];
  const float scale = 0.125f;  // 1/sqrt(64)

  for (int qt = wave; qt < 65; qt += 8) {
    // Q tile, A-layout fragments for K-steps 0..31 and 32..63
    const bf16* qrow = qb + (size_t)(qt * 16 + j) * 64;
    union { v16bf v; v8bf h2[2]; } qa0, qa1;
    qa0.h2[0] = *(const v8bf*)(qrow + 16 * half);
    qa0.h2[1] = *(const v8bf*)(qrow + 8 + 16 * half);
    qa1.h2[0] = *(const v8bf*)(qrow + 32 + 16 * half);
    qa1.h2[1] = *(const v8bf*)(qrow + 40 + 16 * half);

    v8f   O[4] = {};
    float mrun[8], lrun[8];
#pragma unroll
    for (int r = 0; r < 8; ++r) { mrun[r] = -1e30f; lrun[r] = 0.f; }

    for (int kp = 0; kp < 33; ++kp) {
      const int k0 = (2 * kp) * 16;
      const int k1 = k0 + 16;

      // S = Q * K^T for two adjacent key tiles (B-layout: lane j = key row)
      v8f S0 = {}, S1 = {};
      {
        const bf16* kr0 = kb + (size_t)(k0 + j) * 64;
        S0 = wmma_bf16(qa0.v, *(const v16bf*)(kr0 + 16 * half), S0);
        S0 = wmma_bf16(qa1.v, *(const v16bf*)(kr0 + 32 + 16 * half), S0);
        const bf16* kr1 = kb + (size_t)(k1 + j) * 64;
        S1 = wmma_bf16(qa0.v, *(const v16bf*)(kr1 + 16 * half), S1);
        S1 = wmma_bf16(qa1.v, *(const v16bf*)(kr1 + 32 + 16 * half), S1);
      }
      const bool inv0 = (k0 + j) >= 1025;  // key padding mask (per-lane column)
      const bool inv1 = (k1 + j) >= 1025;

      float P0[8], P1[8];
#pragma unroll
      for (int r = 0; r < 8; ++r) {
        float s0 = inv0 ? -1e30f : S0[r] * scale;
        float s1 = inv1 ? -1e30f : S1[r] * scale;
        float t  = fmaxf(s0, s1);
#pragma unroll
        for (int off = 1; off < 16; off <<= 1)
          t = fmaxf(t, __shfl_xor(t, off, 16));
        float mnew = fmaxf(mrun[r], t);
        float corr = __expf(mrun[r] - mnew);
        float p0   = __expf(s0 - mnew);
        float p1   = __expf(s1 - mnew);
        float rs   = p0 + p1;
#pragma unroll
        for (int off = 1; off < 16; off <<= 1)
          rs += __shfl_xor(rs, off, 16);
        lrun[r] = lrun[r] * corr + rs;
        mrun[r] = mnew;
#pragma unroll
        for (int c = 0; c < 4; ++c) O[c][r] *= corr;
        P0[r] = p0; P1[r] = p1;
      }

      // C-layout -> A-layout via per-wave LDS slab (row-major 16x32 bf16)
#pragma unroll
      for (int r = 0; r < 8; ++r) {
        slab[(r + 8 * half) * 32 + j]      = (bf16)P0[r];
        slab[(r + 8 * half) * 32 + 16 + j] = (bf16)P1[r];
      }
      __builtin_amdgcn_wave_barrier();
      union { v16bf v; v8bf h2[2]; } pa;
      const bf16* pr = slab + j * 32;
      pa.h2[0] = *(const v8bf*)(pr + 16 * half);
      pa.h2[1] = *(const v8bf*)(pr + 8 + 16 * half);
      __builtin_amdgcn_wave_barrier();

      // O += P(16x32) * V(32x64); vt rows give contiguous B-layout loads
#pragma unroll
      for (int c = 0; c < 4; ++c) {
        const bf16* vrow = vb + (size_t)(c * 16 + j) * LP + kp * 32 + 16 * half;
        O[c] = wmma_bf16(pa.v, *(const v16bf*)vrow, O[c]);
      }
    }

    // epilogue: normalize, store to [B, L, H*64] bf16
#pragma unroll
    for (int r = 0; r < 8; ++r) {
      const int row = qt * 16 + r + 8 * half;
      if (row < 1025) {
        const float inv = 1.0f / lrun[r];
#pragma unroll
        for (int c = 0; c < 4; ++c) {
          ao[(size_t)(b * 1025 + row) * 1024 + h * 64 + c * 16 + j] =
              (bf16)(O[c][r] * inv);
        }
      }
    }
  }
}

// ---------------------------------------------------------------------------
// Host-side orchestration
// ---------------------------------------------------------------------------
extern "C" void kernel_launch(void* const* d_in, const int* in_sizes, int n_in,
                              void* d_out, int out_size, void* d_ws, size_t ws_size,
                              hipStream_t stream) {
  const float* x      = (const float*)d_in[0];
  const float* sinp   = (const float*)d_in[1];
  const float* cosp   = (const float*)d_in[2];
  const float* qkv_w  = (const float*)d_in[3];
  const float* proj_w = (const float*)d_in[4];
  const float* proj_b = (const float*)d_in[5];
  float*       out    = (float*)d_out;

  const int    B = 16, L = 1025, C = 1024, H = 16, LP = 1056;
  const size_t ML = (size_t)B * L;  // 16400 rows, exactly 1025 16-row tiles

  char* ws    = (char*)d_ws;
  auto  alloc = [&](size_t bytes) -> char* {
    char* p = ws;
    ws += (bytes + 255) & ~(size_t)255;
    return p;
  };
  bf16* xb    = (bf16*)alloc(ML * C * 2);             // x in bf16
  bf16* wqkv  = (bf16*)alloc((size_t)3 * C * C * 2);  // qkv_w bf16
  bf16* wproj = (bf16*)alloc((size_t)C * C * 2);      // proj_w bf16
  bf16* qkv   = (bf16*)alloc(ML * 3 * C * 2);         // qkv GEMM out
  bf16* qh    = (bf16*)alloc((size_t)B * H * LP * 64 * 2);
  bf16* kh    = (bf16*)alloc((size_t)B * H * LP * 64 * 2);
  bf16* vt    = (bf16*)alloc((size_t)B * H * 64 * LP * 2);
  bf16* ao    = (bf16*)alloc(ML * C * 2);             // attention out

  // 1) casts
  {
    int n = (int)(ML * C);
    cast_f32_to_bf16<<<(n + 255) / 256, 256, 0, stream>>>(x, xb, n);
    n = 3 * C * C;
    cast_f32_to_bf16<<<(n + 255) / 256, 256, 0, stream>>>(qkv_w, wqkv, n);
    n = C * C;
    cast_f32_to_bf16<<<(n + 255) / 256, 256, 0, stream>>>(proj_w, wproj, n);
  }
  // 2) qkv = x @ qkv_w^T   (M=16400, N=3072, K=1024)
  gemm_wmma_tdm<true, false><<<dim3(3072 / 512, 1025), 256, 0, stream>>>(
      xb, wqkv, qkv, nullptr, (int)ML, 3 * C, C);
  // 3) RoPE + repack q/k/v
  rope_pack<<<(B * H * LP * 64) / 256, 256, 0, stream>>>(qkv, sinp, cosp, qh, kh, vt);
  // 4) flash attention, one block per (b,h)
  flash_attn<<<B * H, 256, 0, stream>>>(qh, kh, vt, ao);
  // 5) out = ao @ proj_w^T + proj_b   (M=16400, N=1024, K=1024)
  gemm_wmma_tdm<false, true><<<dim3(1024 / 512, 1025), 256, 0, stream>>>(
      ao, wproj, out, proj_b, (int)ML, C, C);
}
